// ScoreBlock_63943473103413
// MI455X (gfx1250) — matrix operand order repaired and checked by reading, not compiled, not run on hardware
//
#include <hip/hip_runtime.h>

// Problem constants (match reference)
#define BB 1024
#define DD 128
#define TT 256
#define HH 64
#define G4H 256  // 4*H
#define NBT (BB / 16)  // 64 batch tiles

typedef __attribute__((ext_vector_type(16))) __bf16 v16bf;
typedef __attribute__((ext_vector_type(8)))  float  v8f;
typedef __attribute__((ext_vector_type(4)))  unsigned int v4u;

// ---------- WMMA fragment loaders (wave32, 16-bit layouts per ISA 7.12.2) ----------
// A (16xK bf16): lane<16 -> row=lane, K = kb+{0..7,16..23}; lane>=16 -> row=lane-16, K = kb+{8..15,24..31}
// `ldrow` = element stride between consecutive logical rows (lets A-tiles cut across [B][T] layouts).
__device__ __forceinline__ v16bf load_afrag(const __bf16* A, size_t ldrow, int row0, int kb, int lane) {
    const __bf16* p = A + (size_t)(row0 + (lane & 15)) * ldrow + kb + ((lane >> 4) << 3);
    v16bf a;
    v4u* pa = (v4u*)&a;
    pa[0] = *(const v4u*)(p);       // K = kb + 0..7 (or 8..15)
    pa[1] = *(const v4u*)(p + 16);  // K = kb + 16..23 (or 24..31)
    return a;
}
// B (Kx16) taken from row-major W[n][k] (B[k,n] = W[n,k]):
// lane<16 -> n=n0+lane, k=kb+{0..7,16..23} ; lane>=16 -> n=n0+lane-16, k=kb+{8..15,24..31}
__device__ __forceinline__ v16bf load_bfrag(const __bf16* W, int ldk, int n0, int kb, int lane) {
    const __bf16* p = W + (size_t)(n0 + (lane & 15)) * ldk + kb + ((lane >> 4) << 3);
    v16bf b;
    v4u* pb = (v4u*)&b;
    pb[0] = *(const v4u*)(p);
    pb[1] = *(const v4u*)(p + 16);
    return b;
}

__device__ __forceinline__ v8f wmma_bf16(v16bf a, v16bf b, v8f c) {
    return __builtin_amdgcn_wmma_f32_16x16x32_bf16(false, a, false, b, (short)0, c, false, false);
}

__device__ __forceinline__ float sigf(float x)  { return 1.0f / (1.0f + __expf(-x)); }
__device__ __forceinline__ float tanhf_(float x){ return 1.0f - 2.0f / (__expf(2.0f * x) + 1.0f); }

// xp fragment index: [t][btile][ntile] fragments of 256 floats, lane-packed (lane*8 + r).
__device__ __forceinline__ size_t xp_frag_off(int t, int btile, int nt, int lane) {
    return ((((size_t)t * NBT + btile) * 16) + nt) * 256 + (size_t)lane * 8;
}

// ---------- small prep kernels ----------
__global__ void k_f32_to_bf16(const float* __restrict__ s, __bf16* __restrict__ d, int n) {
    int i = blockIdx.x * blockDim.x + threadIdx.x;
    if (i < n) d[i] = (__bf16)s[i];
}
__global__ void k_bias_sum(const float* __restrict__ a, const float* __restrict__ b,
                           float* __restrict__ o, int n) {
    int i = blockIdx.x * blockDim.x + threadIdx.x;
    if (i < n) o[i] = a[i] + b[i];
}
// x [B][D][T] f32 -> xt [B][T][D] bf16 (tiled transpose)
__global__ void k_transpose_cvt(const float* __restrict__ x, __bf16* __restrict__ xt) {
    __shared__ float tile[32][33];
    int b  = blockIdx.z;
    int t0 = blockIdx.x * 32;
    int d0 = blockIdx.y * 32;
    for (int i = 0; i < 4; ++i) {
        int d = d0 + threadIdx.y + 8 * i;
        tile[threadIdx.y + 8 * i][threadIdx.x] = x[(size_t)b * DD * TT + (size_t)d * TT + t0 + threadIdx.x];
    }
    __syncthreads();
    for (int i = 0; i < 4; ++i) {
        int t = t0 + threadIdx.y + 8 * i;
        xt[(size_t)b * TT * DD + (size_t)t * DD + d0 + threadIdx.x] =
            (__bf16)tile[threadIdx.x][threadIdx.y + 8 * i];
    }
}

// ---------- input-projection GEMM ----------
// xp_frag(t, btile, nt) = A-tile(rows b0..b0+15 at fixed t) x W^T + bias, stored fragment-packed.
// A: [B][T][K] bf16 (row b,t at offset (b*T+t)*K; row stride within a tile = T*K).
// Grid (btile, t), block = 128 thr (4 waves), wave w owns N-tiles 4w..4w+3.
template <int K>
__global__ void k_in_proj(const __bf16* __restrict__ A, const __bf16* __restrict__ W,
                          const float* __restrict__ bias, float* __restrict__ xpf) {
    int tid   = threadIdx.x;
    int lane  = tid & 31;
    int wave  = tid >> 5;
    int btile = blockIdx.x;
    int t     = blockIdx.y;
    int n_l   = lane & 15;

    const __bf16* Abase = A + (size_t)t * K;  // row b is at Abase + b*(T*K)

    v8f acc[4];
    for (int j = 0; j < 4; ++j) {
        float bv = bias[wave * 64 + j * 16 + n_l];
        for (int r = 0; r < 8; ++r) acc[j][r] = bv;
    }
    for (int kb = 0; kb < K; kb += 32) {
        v16bf a = load_afrag(Abase, (size_t)TT * K, btile * 16, kb, lane);
        for (int j = 0; j < 4; ++j) {
            v16bf b = load_bfrag(W, K, (wave * 4 + j) * 16, kb, lane);
            acc[j] = wmma_bf16(a, b, acc[j]);
        }
    }
    for (int j = 0; j < 4; ++j) {
        float* op = xpf + xp_frag_off(t, btile, wave * 4 + j, lane);
        const float4* f = (const float4*)&acc[j];
        *(float4*)op       = f[0];   // b128 store
        *(float4*)(op + 4) = f[1];   // b128 store
    }
}

// ---------- persistent recurrent LSTM layer ----------
// Grid = B/16 blocks, 256 threads (8 waves). Each block owns 16 batch rows for all T steps.
// W_hh B-fragments preloaded to VGPRs; h in LDS (bf16, A-fragment source); c in registers.
// xp read fragment-packed (b128), next step prefetched (global_prefetch_b8).
template <int LAYER>
__global__ void k_lstm(const float* __restrict__ xpf, const __bf16* __restrict__ Whh,
                       __bf16* __restrict__ ys_out, __bf16* __restrict__ hT_out) {
    __shared__ float  gates[16][G4H];   // 16 KB
    __shared__ __bf16 h_lds[16][HH];    // 2 KB
    int tid   = threadIdx.x;
    int lane  = tid & 31;
    int wave  = tid >> 5;
    int btile = blockIdx.x;
    int b0    = btile * 16;

    // Preload loop-invariant W_hh fragments (wave owns N-tiles 2w, 2w+1; K = 64 -> 2 kfrags)
    v16bf wb[2][2];
    for (int j = 0; j < 2; ++j)
        for (int kk = 0; kk < 2; ++kk)
            wb[j][kk] = load_bfrag(Whh, HH, (wave * 2 + j) * 16, kk * 32, lane);

    // init h = 0
    __bf16* hz = &h_lds[0][0];
    for (int u = tid; u < 16 * HH; u += 256) hz[u] = (__bf16)0.0f;

    // c state in registers: thread -> (erow, ecol..ecol+3)
    int erow = tid >> 4;
    int ecol = (tid & 15) * 4;
    float c0 = 0.f, c1 = 0.f, c2 = 0.f, c3 = 0.f;

    int n_l  = lane & 15;
    int msel = (lane >> 4) * 8;
    int akof = (lane >> 4) << 3;
    int arow = lane & 15;
    __syncthreads();

    for (int t = 0; t < TT; ++t) {
        // Prefetch next step's xp fragments for this wave's tiles
        if (t + 1 < TT) {
            __builtin_prefetch(xpf + xp_frag_off(t + 1, btile, wave * 2, lane), 0, 1);
            __builtin_prefetch(xpf + xp_frag_off(t + 1, btile, wave * 2 + 1, lane), 0, 1);
        }
        // ---- gates = xp[t] + h @ Whh^T (WMMA phase) ----
        v16bf a0, a1;
        {
            const __bf16* p = &h_lds[arow][akof];
            v4u* pa = (v4u*)&a0; pa[0] = *(const v4u*)(p); pa[1] = *(const v4u*)(p + 16);
            const __bf16* q = &h_lds[arow][32 + akof];
            v4u* qa = (v4u*)&a1; qa[0] = *(const v4u*)(q); qa[1] = *(const v4u*)(q + 16);
        }
        for (int j = 0; j < 2; ++j) {
            int nt = wave * 2 + j;
            v8f acc;
            const float* xpp = xpf + xp_frag_off(t, btile, nt, lane);
            float4* f = (float4*)&acc;
            f[0] = *(const float4*)xpp;        // b128 load
            f[1] = *(const float4*)(xpp + 4);  // b128 load
            acc = wmma_bf16(a0, wb[j][0], acc);
            acc = wmma_bf16(a1, wb[j][1], acc);
            int n = nt * 16 + n_l;
            for (int r = 0; r < 8; ++r) gates[msel + r][n] = acc[r];
        }
        __syncthreads();

        // ---- elementwise gate math; PyTorch order i,f,g,o ----
        float4 ig = *(const float4*)&gates[erow][ecol];
        float4 fg = *(const float4*)&gates[erow][64 + ecol];
        float4 gg = *(const float4*)&gates[erow][128 + ecol];
        float4 og = *(const float4*)&gates[erow][192 + ecol];

        c0 = sigf(fg.x) * c0 + sigf(ig.x) * tanhf_(gg.x);
        c1 = sigf(fg.y) * c1 + sigf(ig.y) * tanhf_(gg.y);
        c2 = sigf(fg.z) * c2 + sigf(ig.z) * tanhf_(gg.z);
        c3 = sigf(fg.w) * c3 + sigf(ig.w) * tanhf_(gg.w);
        float h0 = sigf(og.x) * tanhf_(c0);
        float h1 = sigf(og.y) * tanhf_(c1);
        float h2 = sigf(og.z) * tanhf_(c2);
        float h3 = sigf(og.w) * tanhf_(c3);

        __bf16 hb[4];
        hb[0] = (__bf16)h0; hb[1] = (__bf16)h1; hb[2] = (__bf16)h2; hb[3] = (__bf16)h3;
        *(unsigned long long*)&h_lds[erow][ecol] = *(const unsigned long long*)hb;

        if (LAYER == 0) {
            // ys0 [B][T][H] bf16 (feeds layer-1 input projection GEMM)
            __bf16* yp = ys_out + ((size_t)(b0 + erow) * TT + t) * HH + ecol;
            *(unsigned long long*)yp = *(const unsigned long long*)hb;
        } else if (t == TT - 1) {
            __bf16* hp = hT_out + (size_t)(b0 + erow) * HH + ecol;
            *(unsigned long long*)hp = *(const unsigned long long*)hb;
        }
        __syncthreads();
    }
}

// ---------- final linear head: out[b][d] = hT[b]·w_fc[d] + b_fc[d] ----------
// Block = 128 thr (4 waves), wave owns 2 N-tiles (8 total, N=128). Grid = B/16.
__global__ void k_head(const __bf16* __restrict__ hT, const __bf16* __restrict__ Wfc,
                       const float* __restrict__ bfc, float* __restrict__ out) {
    int tid  = threadIdx.x;
    int lane = tid & 31;
    int wave = tid >> 5;
    int row0 = blockIdx.x * 16;
    int n_l  = lane & 15;
    int msel = (lane >> 4) * 8;

    for (int j = 0; j < 2; ++j) {
        int n0 = (wave * 2 + j) * 16;
        v8f acc;
        float bv = bfc[n0 + n_l];
        for (int r = 0; r < 8; ++r) acc[r] = bv;
        for (int kb = 0; kb < HH; kb += 32) {
            v16bf a = load_afrag(hT, HH, row0, kb, lane);
            v16bf b = load_bfrag(Wfc, HH, n0, kb, lane);
            acc = wmma_bf16(a, b, acc);
        }
        for (int r = 0; r < 8; ++r)
            out[(size_t)(row0 + msel + r) * DD + n0 + n_l] = acc[r];
    }
}

extern "C" void kernel_launch(void* const* d_in, const int* in_sizes, int n_in,
                              void* d_out, int out_size, void* d_ws, size_t ws_size,
                              hipStream_t stream) {
    const float* x     = (const float*)d_in[0];
    const float* w_ih0 = (const float*)d_in[1];
    const float* w_hh0 = (const float*)d_in[2];
    const float* b_ih0 = (const float*)d_in[3];
    const float* b_hh0 = (const float*)d_in[4];
    const float* w_ih1 = (const float*)d_in[5];
    const float* w_hh1 = (const float*)d_in[6];
    const float* b_ih1 = (const float*)d_in[7];
    const float* b_hh1 = (const float*)d_in[8];
    const float* w_fc  = (const float*)d_in[9];
    const float* b_fc  = (const float*)d_in[10];
    float* out = (float*)d_out;

    // Workspace layout (~370 MB; xp buffer reused between layers)
    char* ws = (char*)d_ws;
    size_t o = 0;
    auto carve = [&](size_t bytes) { char* p = ws + o; o = (o + bytes + 255) & ~(size_t)255; return p; };
    __bf16* xt_b   = (__bf16*)carve((size_t)BB * TT * DD * 2);   // 67 MB
    float*  xpf    = (float*) carve((size_t)TT * BB * G4H * 4);  // 268 MB (layer0, then layer1)
    __bf16* ys0    = (__bf16*)carve((size_t)BB * TT * HH * 2);   // 33.5 MB
    __bf16* hT     = (__bf16*)carve((size_t)BB * HH * 2);
    __bf16* wih0b  = (__bf16*)carve((size_t)G4H * DD * 2);
    __bf16* whh0b  = (__bf16*)carve((size_t)G4H * HH * 2);
    __bf16* wih1b  = (__bf16*)carve((size_t)G4H * HH * 2);
    __bf16* whh1b  = (__bf16*)carve((size_t)G4H * HH * 2);
    __bf16* wfcb   = (__bf16*)carve((size_t)DD * HH * 2);
    float*  bias0  = (float*) carve(G4H * 4);
    float*  bias1  = (float*) carve(G4H * 4);
    (void)ws_size; (void)in_sizes; (void)n_in; (void)out_size;

    // 1) weight/bias prep
    k_f32_to_bf16<<<(G4H * DD) / 256, 256, 0, stream>>>(w_ih0, wih0b, G4H * DD);
    k_f32_to_bf16<<<(G4H * HH) / 256, 256, 0, stream>>>(w_hh0, whh0b, G4H * HH);
    k_f32_to_bf16<<<(G4H * HH) / 256, 256, 0, stream>>>(w_ih1, wih1b, G4H * HH);
    k_f32_to_bf16<<<(G4H * HH) / 256, 256, 0, stream>>>(w_hh1, whh1b, G4H * HH);
    k_f32_to_bf16<<<(DD * HH) / 256, 256, 0, stream>>>(w_fc, wfcb, DD * HH);
    k_bias_sum<<<1, G4H, 0, stream>>>(b_ih0, b_hh0, bias0, G4H);
    k_bias_sum<<<1, G4H, 0, stream>>>(b_ih1, b_hh1, bias1, G4H);

    // 2) x [B,D,T] -> bf16 [B,T,D]
    {
        dim3 g(TT / 32, DD / 32, BB), b(32, 8);
        k_transpose_cvt<<<g, b, 0, stream>>>(x, xt_b);
    }

    // 3) layer-0 input projection (WMMA GEMM, fragment-packed output), then recurrence
    {
        dim3 g(NBT, TT);
        k_in_proj<DD><<<g, 128, 0, stream>>>(xt_b, wih0b, bias0, xpf);
    }
    k_lstm<0><<<NBT, 256, 0, stream>>>(xpf, whh0b, ys0, nullptr);

    // 4) layer-1 input projection (reuses xpf), then recurrence -> hT
    {
        dim3 g(NBT, TT);
        k_in_proj<HH><<<g, 128, 0, stream>>>(ys0, wih1b, bias1, xpf);
    }
    k_lstm<1><<<NBT, 256, 0, stream>>>(xpf, whh1b, nullptr, hT);

    // 5) linear head
    k_head<<<NBT, 128, 0, stream>>>(hT, wfcb, b_fc, out);
}
